// MyConv2D_70875550318924
// MI455X (gfx1250) — 8-bit matrix operand order repaired and checked by reading, compiled, not run on hardware
//
#include <hip/hip_runtime.h>


typedef int v8i __attribute__((ext_vector_type(8)));
typedef int v4i __attribute__((ext_vector_type(4)));
typedef int v2i __attribute__((ext_vector_type(2)));
typedef unsigned v4u __attribute__((ext_vector_type(4)));

#define HH   112
#define WW   112
#define CIN  64
#define COUT 128
#define NBATCH 32

#define WSEGS 7
#define STRIPS (NBATCH * HH * WSEGS)
#define CONV_BLOCKS (STRIPS / 8)

#define QW_DWORDS (9 * 8 * 32 * 8)
#define QX_BYTES  ((size_t)NBATCH * HH * WW * CIN)
#define QX_DWORDS (QX_BYTES / 4)

__device__ __forceinline__ int quant8(float x) {
    float r = rintf(x * 128.0f);
    r = fminf(fmaxf(r, -128.0f), 127.0f);
    return (int)r;
}

__device__ __forceinline__ unsigned pack4(int b0, int b1, int b2, int b3) {
    return (unsigned)(b0 & 0xff) | ((unsigned)(b1 & 0xff) << 8) |
           ((unsigned)(b2 & 0xff) << 16) | ((unsigned)(b3 & 0xff) << 24);
}

__global__ __launch_bounds__(256) void quant_input_kernel(
    const float* __restrict__ x, unsigned* __restrict__ qx, int ndw)
{
    int t = blockIdx.x * 256 + threadIdx.x;
    if (t >= ndw) return;
    float4 v = ((const float4*)x)[t];
    qx[t] = pack4(quant8(v.x), quant8(v.y), quant8(v.z), quant8(v.w));
}

__global__ __launch_bounds__(256) void quant_weight_pack_kernel(
    const float* __restrict__ w, unsigned* __restrict__ qw)
{
    int t = blockIdx.x * 256 + threadIdx.x;
    if (t >= QW_DWORDS) return;
    int v    = t & 7;
    int lane = (t >> 3) & 31;
    int nt   = (t >> 8) & 7;
    int tap  = t >> 11;
    int co   = nt * 16 + (lane & 15);
    int half = lane >> 4;
    int kbase = (v >> 1) * 16 + 8 * half + 4 * (v & 1);
    int b[4];
    #pragma unroll
    for (int j = 0; j < 4; j++) {
        int ci = kbase + j;
        b[j] = quant8(w[((size_t)tap * CIN + ci) * COUT + co]);
    }
    qw[t] = pack4(b[0], b[1], b[2], b[3]);
}

__global__ __launch_bounds__(256) void conv_wmma_kernel(
    const unsigned char* __restrict__ qx, const unsigned* __restrict__ qw,
    float* __restrict__ out)
{
    __shared__ unsigned lds_w[QW_DWORDS];
    __shared__ unsigned lds_o[8 * 2048];

    int tid  = threadIdx.x;
    int wave = tid >> 5;
    int lane = tid & 31;

    if (wave == 0) {
        unsigned lds_addr = (unsigned)(size_t)(void*)lds_w;
        unsigned long long ga = (unsigned long long)(size_t)qw;
        v4u g0 = { 1u,
                   lds_addr,
                   (unsigned)(ga & 0xffffffffu),
                   (unsigned)((ga >> 32) & 0x01ffffffu) | 0x80000000u };
        v8i g1 = { 0x20000,
                   (int)0x48000000,
                   0x10000,
                   (int)0x48000000,
                   1,
                   0x4800,
                   0, 0 };
        v4i gz  = { 0, 0, 0, 0 };
        v8i gz8 = { 0, 0, 0, 0, 0, 0, 0, 0 };
        __builtin_amdgcn_tensor_load_to_lds(g0, g1, gz, gz, gz8, 0);
        __builtin_amdgcn_s_wait_tensorcnt(0);
    }
    __syncthreads();

    int s      = blockIdx.x * 8 + wave;
    int w_base = (s % WSEGS) * 16;
    int h      = (s / WSEGS) % HH;
    int n      = s / (WSEGS * HH);

    int half = lane >> 4;
    int mrow = lane & 15;
    int wpix = w_base + mrow;

    v8i acc[8];
    #pragma unroll
    for (int i = 0; i < 8; i++) acc[i] = v8i{0, 0, 0, 0, 0, 0, 0, 0};

    #pragma unroll
    for (int kh = 0; kh < 3; kh++) {
        int ih = h + kh - 1;
        bool vh = (ih >= 0) && (ih < HH);
        #pragma unroll
        for (int kw = 0; kw < 3; kw++) {
            int tap = kh * 3 + kw;
            int iw  = wpix + kw - 1;
            bool valid = vh && (iw >= 0) && (iw < WW);

            v2i c0{}, c1{}, c2{}, c3{};
            if (valid) {
                const unsigned char* p =
                    qx + (((size_t)(n * HH + ih) * WW + iw) * CIN) + half * 8;
                c0 = *(const v2i*)(p);
                c1 = *(const v2i*)(p + 16);
                c2 = *(const v2i*)(p + 32);
                c3 = *(const v2i*)(p + 48);
            }
            v8i a;
            a[0] = c0[0]; a[1] = c0[1]; a[2] = c1[0]; a[3] = c1[1];
            a[4] = c2[0]; a[5] = c2[1]; a[6] = c3[0]; a[7] = c3[1];

            #pragma unroll
            for (int nt = 0; nt < 8; nt++) {
                const unsigned* bp = &lds_w[(tap * 8 + nt) * 256 + lane * 8];
                v4i b0 = *(const v4i*)(bp);
                v4i b1 = *(const v4i*)(bp + 4);
                v8i b;
                b[0] = b0[0]; b[1] = b0[1]; b[2] = b0[2]; b[3] = b0[3];
                b[4] = b1[0]; b[5] = b1[1]; b[6] = b1[2]; b[7] = b1[3];
                acc[nt] = __builtin_amdgcn_wmma_i32_16x16x64_iu8(
                    true, a, true, b, acc[nt], false, false);
            }
        }
    }

    unsigned* stag = &lds_o[wave * 2048];
    #pragma unroll
    for (int nt = 0; nt < 8; nt++) {
        int ch = nt * 16 + (lane & 15);
        #pragma unroll
        for (int r = 0; r < 8; r++) {
            int pix = r + 8 * half;
            float q = (float)(acc[nt][r] >> 7) * 0.0078125f;
            stag[pix * COUT + ch] = __float_as_uint(q);
        }
    }
    float* outp = out + ((size_t)(n * HH + h) * WW + w_base) * COUT;
    #pragma unroll
    for (int i = 0; i < 16; i++) {
        int off = (i * 32 + lane) * 4;
        v4i d = *(const v4i*)&stag[off];
        *(v4i*)&outp[off] = d;
    }
}

extern "C" void kernel_launch(void* const* d_in, const int* in_sizes, int n_in,
                              void* d_out, int out_size, void* d_ws, size_t ws_size,
                              hipStream_t stream) {
    const float* x = (const float*)d_in[0];
    const float* k = (const float*)d_in[1];
    float* out = (float*)d_out;

    unsigned* qw = (unsigned*)d_ws;
    unsigned char* qx = (unsigned char*)d_ws + QW_DWORDS * 4;

    quant_weight_pack_kernel<<<QW_DWORDS / 256, 256, 0, stream>>>(k, qw);
    quant_input_kernel<<<QX_DWORDS / 256, 256, 0, stream>>>(x, (unsigned*)qx,
                                                            (int)QX_DWORDS);
    conv_wmma_kernel<<<CONV_BLOCKS, 256, 0, stream>>>(qx, qw, out);
}
